// GGNNMessageLayer_25194278158854
// MI455X (gfx1250) — matrix operand order repaired and checked by reading, not compiled, not run on hardware
//
#include <hip/hip_runtime.h>

// ---- vector types matching CDNA5 WMMA operand layouts (wave32) ----
typedef __attribute__((ext_vector_type(16))) __bf16 bf16x16;
typedef __attribute__((ext_vector_type(8)))  __bf16 bf16x8;
typedef __attribute__((ext_vector_type(4)))  __bf16 bf16x4;
typedef __attribute__((ext_vector_type(8)))  float  floatx8;

#define THREADS 256
#define WAVES_PER_BLOCK 8   // wave32
#define TILE_COLS 4         // 16x(4*16) output strip per wave
#define LDS_STRIDE 520      // 512 + 8 pad: row-to-row = 260 dwords = 4 banks offset -> conflict-free

// ---------------- zero fill (accumulators must be zeroed every launch) ----------------
__global__ __launch_bounds__(THREADS) void zero_f4(float4* __restrict__ p, int n4) {
    int i = blockIdx.x * blockDim.x + threadIdx.x;
    if (i < n4) p[i] = make_float4(0.f, 0.f, 0.f, 0.f);
}

// ---------------- f32 -> bf16 conversion (4 elems/thread) ----------------
__global__ __launch_bounds__(THREADS) void cvt_f32_to_bf16_x4(const float* __restrict__ in,
                                                              __bf16* __restrict__ out, int n4) {
    int i = blockIdx.x * blockDim.x + threadIdx.x;
    if (i >= n4) return;
    float4 f = reinterpret_cast<const float4*>(in)[i];
    bf16x4 o;
    o.x = (__bf16)f.x; o.y = (__bf16)f.y; o.z = (__bf16)f.z; o.w = (__bf16)f.w;
    reinterpret_cast<bf16x4*>(out)[i] = o;
}

// ---------------- WMMA GEMM: C[n, j] = sum_k A[n,k] * W[j,k] + b[j] ----------------
// One block = one 16-row A tile (staged in LDS, shared by 8 waves) x 8 column groups.
// Each wave: 16 x 64 output strip; K fully unrolled in steps of 32 with register
// double-buffering so B (global/L2) and A (LDS) fragments are fetched one k-step ahead.
__global__ __launch_bounds__(THREADS)
void gemm_bf16_wmma(const __bf16* __restrict__ A, const __bf16* __restrict__ Wb,
                    const float* __restrict__ bias, __bf16* __restrict__ C,
                    int nRows, int nCols, int K) {
    __shared__ __bf16 Alds[16 * LDS_STRIDE];   // 16 rows x 512 K (padded), 16.3 KB

    const int lane = threadIdx.x & 31;
    const int wave = threadIdx.x >> 5;

    const int blocksPerRow = nCols / (16 * TILE_COLS * WAVES_PER_BLOCK); // 2048/512 = 4
    const int rowTile  = blockIdx.x / blocksPerRow;
    const int colGroup = (blockIdx.x % blocksPerRow) * WAVES_PER_BLOCK + wave;
    if (rowTile * 16 >= nRows) return;          // uniform across block

    const int row0 = rowTile * 16;
    const int col0 = colGroup * 16 * TILE_COLS;

    // ---- cooperative stage of A tile: 256 threads x 32 elems (4x b128 each) ----
    {
        const int r  = threadIdx.x >> 4;        // 0..15 row
        const int kb = (threadIdx.x & 15) * 32; // 0..480 K base
        const __bf16* src = A + (size_t)(row0 + r) * K + kb;
        __bf16* dst = Alds + r * LDS_STRIDE + kb;
        *reinterpret_cast<bf16x8*>(dst +  0) = *reinterpret_cast<const bf16x8*>(src +  0);
        *reinterpret_cast<bf16x8*>(dst +  8) = *reinterpret_cast<const bf16x8*>(src +  8);
        *reinterpret_cast<bf16x8*>(dst + 16) = *reinterpret_cast<const bf16x8*>(src + 16);
        *reinterpret_cast<bf16x8*>(dst + 24) = *reinterpret_cast<const bf16x8*>(src + 24);
    }
    __syncthreads();

    const int m    = lane & 15;   // row (A) / col (B) owned by this lane
    const int half = lane >> 4;   // K-half selector: k = 16*half + s (this code's contiguous order)
    const __bf16* __restrict__ arow = Alds + m * LDS_STRIDE + half * 16;

    // per-tile B pointers: column (col0+t*16+m), lane's K-half offset folded in
    const __bf16* bptr[TILE_COLS];
#pragma unroll
    for (int t = 0; t < TILE_COLS; ++t)
        bptr[t] = Wb + (size_t)(col0 + t * 16 + m) * K + half * 16;

    floatx8 acc[TILE_COLS];
#pragma unroll
    for (int t = 0; t < TILE_COLS; ++t) acc[t] = (floatx8){0,0,0,0,0,0,0,0};

    union Afrag { bf16x8 h[2]; bf16x16 v; };

    // prologue: fetch k0 = 0 fragments
    Afrag acur;
    acur.h[0] = *reinterpret_cast<const bf16x8*>(arow);
    acur.h[1] = *reinterpret_cast<const bf16x8*>(arow + 8);
    bf16x16 bcur[TILE_COLS];
#pragma unroll
    for (int t = 0; t < TILE_COLS; ++t)
        bcur[t] = *reinterpret_cast<const bf16x16*>(bptr[t]);

#pragma unroll
    for (int k0 = 0; k0 < 512; k0 += 32) {
        const int kn = k0 + 32;
        Afrag anext;
        bf16x16 bnext[TILE_COLS];
        if (kn < 512) {   // compile-time resolved (full unroll)
            anext.h[0] = *reinterpret_cast<const bf16x8*>(arow + kn);
            anext.h[1] = *reinterpret_cast<const bf16x8*>(arow + kn + 8);
#pragma unroll
            for (int t = 0; t < TILE_COLS; ++t)
                bnext[t] = *reinterpret_cast<const bf16x16*>(bptr[t] + kn);
        }
#pragma unroll
        for (int t = 0; t < TILE_COLS; ++t)
            acc[t] = __builtin_amdgcn_wmma_f32_16x16x32_bf16(
                false, acur.v, false, bcur[t], (short)0, acc[t], false, false);
        if (kn < 512) {
            acur = anext;
#pragma unroll
            for (int t = 0; t < TILE_COLS; ++t) bcur[t] = bnext[t];
        }
    }

    // D layout: VGPR v holds row (v + 8*half), col = lane&15
#pragma unroll
    for (int t = 0; t < TILE_COLS; ++t) {
        const int j = col0 + t * 16 + m;
        const float bj = bias[j];
#pragma unroll
        for (int v = 0; v < 8; ++v) {
            const int r = row0 + v + 8 * half;
            C[(size_t)r * nCols + j] = (__bf16)(acc[t][v] + bj);
        }
    }
}

// ---------------- edge scatter: one wave per edge ----------------
// msg = prop[src, e*512 .. e*512+511]; out[tgt] += msg; counts[tgt] += 1
__global__ __launch_bounds__(THREADS)
void scatter_edges(const long long* __restrict__ edges, const __bf16* __restrict__ prop,
                   float* __restrict__ out, float* __restrict__ counts,
                   int nEdgesTotal, int M, int D, int ED) {
    const int gw = (int)((blockIdx.x * blockDim.x + threadIdx.x) >> 5);
    if (gw >= nEdgesTotal) return;
    const int lane = threadIdx.x & 31;

    // uniform per wave -> scalar loads
    const long long src = edges[(size_t)2 * gw];
    const long long tgt = edges[(size_t)2 * gw + 1];
    const int e = gw / M;

    const __bf16* __restrict__ p = prop + (size_t)src * ED + (size_t)e * D + lane * 16;
    float* __restrict__ o = out + (size_t)tgt * D + lane * 16;

    bf16x8 lo = *reinterpret_cast<const bf16x8*>(p);
    bf16x8 hi = *reinterpret_cast<const bf16x8*>(p + 8);
#pragma unroll
    for (int j = 0; j < 8; ++j) atomicAdd(o + j, (float)lo[j]);
#pragma unroll
    for (int j = 0; j < 8; ++j) atomicAdd(o + 8 + j, (float)hi[j]);

    if (lane == 0) atomicAdd(counts + tgt, 1.0f);
}

// ---------------- finalize: out = out / max(count,1) + 1e-8 ----------------
__global__ __launch_bounds__(THREADS)
void finalize_div(float* __restrict__ out, const float* __restrict__ counts, int n4, int D) {
    int i = blockIdx.x * blockDim.x + threadIdx.x;
    if (i >= n4) return;
    const int idx = i * 4;
    const float c = counts[idx / D];
    const float inv = 1.0f / ((c == 0.0f) ? 1.0f : c);
    float4 v = reinterpret_cast<const float4*>(out)[i];
    v.x = v.x * inv + 1e-8f; v.y = v.y * inv + 1e-8f;
    v.z = v.z * inv + 1e-8f; v.w = v.w * inv + 1e-8f;
    reinterpret_cast<float4*>(out)[i] = v;
}

static inline size_t align256(size_t x) { return (x + 255) & ~(size_t)255; }

extern "C" void kernel_launch(void* const* d_in, const int* in_sizes, int n_in,
                              void* d_out, int out_size, void* d_ws, size_t ws_size,
                              hipStream_t stream) {
    const float*     node_states = (const float*)d_in[0];      // (N, 512) f32
    const long long* edges       = (const long long*)d_in[1];  // (E, M, 2) i64
    const float*     W           = (const float*)d_in[2];      // (E*512, 512) f32
    const float*     bvec        = (const float*)d_in[3];      // (E*512,) f32
    float*           out         = (float*)d_out;              // (N, 512) f32

    const int D  = 512;
    const int N  = in_sizes[0] / D;          // 50000
    const int ED = in_sizes[3];              // 2048
    const int E  = ED / D;                   // 4
    const int M  = in_sizes[1] / (2 * E);    // 125000
    const int nEdges = E * M;                // 500000

    // workspace layout (bf16 stored as __bf16)
    char* ws = (char*)d_ws;
    const size_t off_counts = 0;
    const size_t off_A      = align256((size_t)N * sizeof(float));
    const size_t off_W      = off_A + align256((size_t)N * D * sizeof(__bf16));
    const size_t off_prop   = off_W + align256((size_t)ED * D * sizeof(__bf16));
    float*  counts  = (float*)(ws + off_counts);
    __bf16* A_bf    = (__bf16*)(ws + off_A);
    __bf16* W_bf    = (__bf16*)(ws + off_W);
    __bf16* prop_bf = (__bf16*)(ws + off_prop);
    (void)ws_size; (void)n_in;

    // 1) zero accumulators (harness does not re-zero between replays)
    {
        int n4 = out_size / 4;
        zero_f4<<<(n4 + THREADS - 1) / THREADS, THREADS, 0, stream>>>((float4*)out, n4);
        int c4 = N / 4;
        zero_f4<<<(c4 + THREADS - 1) / THREADS, THREADS, 0, stream>>>((float4*)counts, c4);
    }

    // 2) convert node_states and W to bf16
    {
        int a4 = (N * D) / 4;
        cvt_f32_to_bf16_x4<<<(a4 + THREADS - 1) / THREADS, THREADS, 0, stream>>>(node_states, A_bf, a4);
        int w4 = (ED * D) / 4;
        cvt_f32_to_bf16_x4<<<(w4 + THREADS - 1) / THREADS, THREADS, 0, stream>>>(W, W_bf, w4);
    }

    // 3) WMMA GEMM: prop (N x 2048) bf16; one block per (row tile x 8 col groups)
    {
        const int rowTiles     = (N + 15) / 16;                           // 3125
        const int blocksPerRow = ED / (16 * TILE_COLS * WAVES_PER_BLOCK); // 4
        const int blocks       = rowTiles * blocksPerRow;                 // 12500
        gemm_bf16_wmma<<<blocks, THREADS, 0, stream>>>(A_bf, W_bf, bvec, prop_bf, N, ED, D);
    }

    // 4) scatter messages with f32 atomics, counts on lane 0
    {
        const long long waves = nEdges;
        const int blocks = (int)((waves + WAVES_PER_BLOCK - 1) / WAVES_PER_BLOCK);
        scatter_edges<<<blocks, THREADS, 0, stream>>>(edges, prop_bf, out, counts, nEdges, M, D, ED);
    }

    // 5) divide by counts and add epsilon
    {
        int n4 = out_size / 4;
        finalize_div<<<(n4 + THREADS - 1) / THREADS, THREADS, 0, stream>>>(out, counts, n4, D);
    }
}
